// HungarianMatcher_80436147520093
// MI455X (gfx1250) — matrix operand order repaired and checked by reading, not compiled, not run on hardware
//
#include <hip/hip_runtime.h>

typedef __attribute__((ext_vector_type(16))) _Float16 v16h;
typedef __attribute__((ext_vector_type(8)))  float    v8f;

#define BN     9600          // b*n = 32*300
#define NC     91            // classes
#define MT     1280          // targets
#define TQ     (BN / 16)     // 600 query tiles
#define TT     (MT / 16)     // 80 target tiles
#define NTILES (TQ * TT)     // 48000 tiles, one wave32 each

__global__ __launch_bounds__(256)
void detr_cost_kernel(const float*  __restrict__ scores,   // [9600, 91]
                      const float4* __restrict__ qboxes,   // [9600] cxcywh
                      const int*    __restrict__ labels,   // [1280]
                      const float4* __restrict__ tboxes,   // [1280] cxcywh
                      float*        __restrict__ out)      // [9600, 1280]
{
    const int lane = threadIdx.x & 31;
    const int wave = threadIdx.x >> 5;
    const int tile = blockIdx.x * 8 + wave;
    if (tile >= NTILES) return;              // uniform per wave; EXEC stays all-1s

    const int tq    = tile / TT;
    const int tt    = tile - tq * TT;
    const int qbase = tq * 16;
    const int tbase = tt * 16;

    const int  m   = lane & 15;
    const bool hiL = lane >= 16;

    // ---- cost_class tile via exact one-hot GEMM: (-scores) x OneHot ----
    const int    myLabel = labels[tbase + m];                 // B column = lane%16
    const float* srow    = scores + (size_t)(qbase + m) * NC; // A row    = lane%16

    v8f acc = {};
    #pragma unroll
    for (int c = 0; c < 3; ++c) {            // K = 91 padded to 96, 3 chunks of 32
        const int k0  = c * 32;
        const int kbA = k0 + (hiL ? 8 : 0);  // A: halves 0-7 -> kbA+i, 8-15 -> kbA+16+i
        const int kbB = k0 + (hiL ? 8 : 0);  // B: halves 0-7 -> kbB+i, 8-15 -> kbB+16+(i-8)
        v16h ahi, alo, bb;
        #pragma unroll
        for (int i = 0; i < 8; ++i) {
            const int ka = kbA + i;
            const int kc = kbA + 16 + i;
            // zero-pad beyond class 90 (garbage->f16 could be NaN; NaN*0 = NaN in WMMA)
            const float s0 = (ka < NC) ? -srow[ka] : 0.0f;
            const float s1 = (kc < NC) ? -srow[kc] : 0.0f;
            const _Float16 h0 = (_Float16)s0;
            const _Float16 h1 = (_Float16)s1;
            ahi[i]     = h0;  alo[i]     = (_Float16)(s0 - (float)h0);
            ahi[8 + i] = h1;  alo[8 + i] = (_Float16)(s1 - (float)h1);
        }
        #pragma unroll
        for (int i = 0; i < 16; ++i)
            bb[i] = (myLabel == (kbB + i + (i >= 8 ? 8 : 0))) ? (_Float16)1.0f : (_Float16)0.0f;

        // split-f16 double accumulation: exact to ~2^-22 relative
        acc = __builtin_amdgcn_wmma_f32_16x16x32_f16(false, ahi, false, bb,
                                                     (short)0, acc, false, false);
        acc = __builtin_amdgcn_wmma_f32_16x16x32_f16(false, alo, false, bb,
                                                     (short)0, acc, false, false);
    }
    // acc[v] = -prob[qbase + v + 8*hiL, label[tbase + lane%16]]

    // ---- L1 + GIoU terms: lane owns one target column, 8 query rows ----
    const int    t  = tbase + m;
    const float4 tb = tboxes[t];
    const float tx1 = tb.x - 0.5f * tb.z, ty1 = tb.y - 0.5f * tb.w;
    const float tx2 = tb.x + 0.5f * tb.z, ty2 = tb.y + 0.5f * tb.w;
    const float tarea = (tx2 - tx1) * (ty2 - ty1);
    const int   roff  = hiL ? 8 : 0;

    #pragma unroll
    for (int v = 0; v < 8; ++v) {
        const int    q  = qbase + roff + v;
        const float4 qb = qboxes[q];

        const float l1 = fabsf(qb.x - tb.x) + fabsf(qb.y - tb.y)
                       + fabsf(qb.z - tb.z) + fabsf(qb.w - tb.w);

        const float qx1 = qb.x - 0.5f * qb.z, qy1 = qb.y - 0.5f * qb.w;
        const float qx2 = qb.x + 0.5f * qb.z, qy2 = qb.y + 0.5f * qb.w;
        const float qarea = (qx2 - qx1) * (qy2 - qy1);

        float iw = fminf(qx2, tx2) - fmaxf(qx1, tx1);
        float ih = fminf(qy2, ty2) - fmaxf(qy1, ty1);
        iw = fmaxf(iw, 0.0f);  ih = fmaxf(ih, 0.0f);
        const float inter = iw * ih;
        const float uni   = qarea + tarea - inter;
        const float iou   = inter / uni;

        float ew = fmaxf(qx2, tx2) - fminf(qx1, tx1);
        float eh = fmaxf(qy2, ty2) - fminf(qy1, ty1);
        ew = fmaxf(ew, 0.0f);  eh = fmaxf(eh, 0.0f);
        const float earea = ew * eh;
        const float giou  = iou - (earea - uni) / earea;

        out[(size_t)q * MT + t] = 5.0f * l1 + acc[v] - 2.0f * giou;
    }
}

extern "C" void kernel_launch(void* const* d_in, const int* in_sizes, int n_in,
                              void* d_out, int out_size, void* d_ws, size_t ws_size,
                              hipStream_t stream) {
    const float*  scores = (const float*) d_in[0];   // [32,300,91] f32
    const float4* qboxes = (const float4*)d_in[1];   // [32,300,4]  f32
    const int*    labels = (const int*)   d_in[2];   // [1280]      int
    const float4* tboxes = (const float4*)d_in[3];   // [1280,4]    f32
    float*        out    = (float*)d_out;            // [32,300,1280] f32

    dim3 grid(NTILES / 8);   // 48000 tiles / 8 waves per block = 6000 blocks
    dim3 block(256);         // 8 wave32s
    hipLaunchKernelGGL(detr_cost_kernel, grid, block, 0, stream,
                       scores, qboxes, labels, tboxes, out);
}
